// PatternBasedV2_75213467288149
// MI455X (gfx1250) — matrix operand order repaired and checked by reading, compiled, not run on hardware
//
#include <hip/hip_runtime.h>
#include <stdint.h>

typedef __attribute__((ext_vector_type(16))) __bf16          v16bf;
typedef __attribute__((ext_vector_type(8)))  float           v8f;
typedef __attribute__((ext_vector_type(16))) unsigned short  v16u;
typedef __attribute__((ext_vector_type(8)))  unsigned short  v8u;

union Frag   { v16bf b; v16u u; uint32_t d[8]; };
union BS     { __bf16 h; unsigned short u; };

#define NCOLS 65536   // B * 8 symmetries
#define NGRP  16

// Packed bf16 ReLU: one VOP3P op per dword (2 bf16 lanes).
__device__ __forceinline__ uint32_t relu_pk_bf16(uint32_t x) {
  uint32_t r;
  asm volatile("v_pk_max_num_bf16 %0, %1, 0" : "=v"(r) : "v"(x));
  return r;
}

// ---------------------------------------------------------------------------
// f32 -> bf16 weight conversion
// ---------------------------------------------------------------------------
__global__ void cvt_bf16_kernel(const float* __restrict__ src,
                                unsigned short* __restrict__ dst, int n) {
  for (int i = blockIdx.x * blockDim.x + threadIdx.x; i < n;
       i += gridDim.x * blockDim.x) {
    BS cv; cv.h = (__bf16)src[i]; dst[i] = cv.u;
  }
}

// ---------------------------------------------------------------------------
// Pattern embedding: x [B,2,8,8] -> m as bf16 [16][B*8][64]
// ---------------------------------------------------------------------------
__device__ const unsigned long long BASE_BITS_DEV[16] = {
    17151ull, 65280ull, 16711680ull, 4278190080ull, 7967ull, 16975631ull,
    4345695256ull, 1108169199648ull, 283691315109952ull, 72624976668147840ull,
    15549ull, 202245891ull, 460551ull, 9471ull, 168169219ull, 8623628035ull};

struct EmbedPtrs { const float* w[16]; };

__global__ __launch_bounds__(256) void embed_kernel(
    const float* __restrict__ x, EmbedPtrs ew, __bf16* __restrict__ out) {
  __shared__ float xs[128];          // [2][8][8]
  __shared__ float vxs[8][2][64];    // symmetry-expanded boards
  __shared__ int plist[16][16];
  __shared__ int plen[16];
  const int b = blockIdx.x;
  const int t = threadIdx.x;
  if (t < 128) xs[t] = x[(long)b * 128 + t];
  if (t < 16) {
    unsigned long long bits = BASE_BITS_DEV[t];
    int n = 0;
    for (int i = 0; i < 64; ++i)
      if ((bits >> i) & 1ull) plist[t][n++] = i;
    plen[t] = n;
  }
  __syncthreads();
  for (int idx = t; idx < 1024; idx += 256) {
    int s = idx >> 7, ch = (idx >> 6) & 1, p = idx & 63;
    int r = p >> 3, c = p & 7;
    int r2 = (s & 2) ? 7 - r : r;
    int c2 = (s & 4) ? 7 - c : c;
    float v = (s & 1) ? xs[ch * 64 + c2 * 8 + r2]
                      : xs[ch * 64 + r2 * 8 + c2];
    vxs[s][ch][p] = v;
  }
  __syncthreads();
  for (int o = t; o < 8192; o += 256) {           // 16 groups * 8 syms * 64 feats
    int g = o >> 9, s = (o >> 6) & 7, f = o & 63;
    int P = plen[g];
    const float* wr = ew.w[g] + (long)f * (2 * P);
    float acc = 0.f;
    for (int j = 0; j < P; ++j) {
      int p = plist[g][j];
      acc += vxs[s][0][p] * wr[j] + vxs[s][1][p] * wr[P + j];
    }
    long col = (long)b * 8 + s;
    BS cv; cv.h = (__bf16)acc;
    ((unsigned short*)out)[((long)g * NCOLS + col) * 64 + f] = cv.u;
  }
}

// ---------------------------------------------------------------------------
// Grouped 1x1 conv, Cout=Cin=64, relu on input, optional residual add.
// in/out/resid: bf16 [16][NCOLS][64]; w: bf16 [16][64][64]; bias: f32 [16][64]
// grid: (NCOLS/128, 16), block 256 (8 waves); wave w owns one 16-col slab.
// ---------------------------------------------------------------------------
__global__ __launch_bounds__(256) void conv64_wmma_kernel(
    const __bf16* __restrict__ in, __bf16* __restrict__ out,
    const __bf16* __restrict__ w, const float* __restrict__ bias,
    const __bf16* __restrict__ resid) {
  __shared__ unsigned short wlds[64 * 64];
  const int g = blockIdx.y;
  const long colbase = (long)blockIdx.x * 128;

  { // stage group weights into LDS (8 KB)
    const unsigned int* wsrc = (const unsigned int*)(w + (long)g * 4096);
    unsigned int* wdst = (unsigned int*)wlds;
    for (int i = threadIdx.x; i < 2048; i += 256) wdst[i] = wsrc[i];
  }
  __syncthreads();

  const int lane = threadIdx.x & 31;
  const int wave = threadIdx.x >> 5;
  const int half = lane >> 4;
  const int mrow = lane & 15;
  const long col = colbase + wave * 16 + mrow;
  const __bf16* incol = in + ((long)g * NCOLS + col) * 64;

  // B fragments (activation columns), packed-bf16 relu applied.
  // B layout: lane half h holds K = kk*32 + h*16 + e (16 contiguous).
  Frag bfrag[2];
#pragma unroll
  for (int kk = 0; kk < 2; ++kk) {
    const uint4* p = (const uint4*)(incol + kk * 32 + half * 16);
    uint4 lo = p[0], hi = p[1];
    bfrag[kk].d[0] = relu_pk_bf16(lo.x);
    bfrag[kk].d[1] = relu_pk_bf16(lo.y);
    bfrag[kk].d[2] = relu_pk_bf16(lo.z);
    bfrag[kk].d[3] = relu_pk_bf16(lo.w);
    bfrag[kk].d[4] = relu_pk_bf16(hi.x);
    bfrag[kk].d[5] = relu_pk_bf16(hi.y);
    bfrag[kk].d[6] = relu_pk_bf16(hi.z);
    bfrag[kk].d[7] = relu_pk_bf16(hi.w);
  }

#pragma unroll
  for (int MT = 0; MT < 4; ++MT) {
    const int row = MT * 16 + mrow;
    v8f c = {};
#pragma unroll
    for (int kk = 0; kk < 2; ++kk) {
      // A layout: K = kk*32 + half*16 + e (16 contiguous)
      const uint4* ap = (const uint4*)(wlds + row * 64 + kk * 32 + half * 16);
      uint4 alo = ap[0], ahi = ap[1];
      Frag af;
      af.d[0] = alo.x; af.d[1] = alo.y; af.d[2] = alo.z; af.d[3] = alo.w;
      af.d[4] = ahi.x; af.d[5] = ahi.y; af.d[6] = ahi.z; af.d[7] = ahi.w;
      c = __builtin_amdgcn_wmma_f32_16x16x32_bf16(false, af.b, false, bfrag[kk].b,
                                                  (short)0, c, false, false);
    }
    const int fo = MT * 16 + half * 8;           // 8 contiguous output features
    const float* bp = bias + g * 64 + fo;
    float rv[8];
#pragma unroll
    for (int v = 0; v < 8; ++v) rv[v] = c[v] + bp[v];
    if (resid) {
      v8u rr = *(const v8u*)(resid + ((long)g * NCOLS + col) * 64 + fo);
#pragma unroll
      for (int v = 0; v < 8; ++v) { BS cv; cv.u = rr[v]; rv[v] += (float)cv.h; }
    }
    v8u st;
#pragma unroll
    for (int v = 0; v < 8; ++v) { BS cv; cv.h = (__bf16)rv[v]; st[v] = cv.u; }
    *(v8u*)(out + ((long)g * NCOLS + col) * 64 + fo) = st;
  }
}

// ---------------------------------------------------------------------------
// Last conv Cout=128 fused with sum over (group, symmetry):
// b5[b][fo] += sum_s out[g][b*8+s][fo]  via shfl-reduce + atomicAdd.
// ---------------------------------------------------------------------------
__global__ __launch_bounds__(256) void lastconv_wmma_kernel(
    const __bf16* __restrict__ in, const __bf16* __restrict__ w,
    const float* __restrict__ bias, float* __restrict__ b5) {
  __shared__ unsigned short wlds[128 * 64];
  const int g = blockIdx.y;
  const long colbase = (long)blockIdx.x * 128;

  { const unsigned int* wsrc = (const unsigned int*)(w + (long)g * 8192);
    unsigned int* wdst = (unsigned int*)wlds;
    for (int i = threadIdx.x; i < 4096; i += 256) wdst[i] = wsrc[i]; }
  __syncthreads();

  const int lane = threadIdx.x & 31;
  const int wave = threadIdx.x >> 5;
  const int half = lane >> 4;
  const int mrow = lane & 15;
  const long col = colbase + wave * 16 + mrow;
  const __bf16* incol = in + ((long)g * NCOLS + col) * 64;

  Frag bfrag[2];
#pragma unroll
  for (int kk = 0; kk < 2; ++kk) {
    const uint4* p = (const uint4*)(incol + kk * 32 + half * 16);
    uint4 lo = p[0], hi = p[1];
    bfrag[kk].d[0] = relu_pk_bf16(lo.x);
    bfrag[kk].d[1] = relu_pk_bf16(lo.y);
    bfrag[kk].d[2] = relu_pk_bf16(lo.z);
    bfrag[kk].d[3] = relu_pk_bf16(lo.w);
    bfrag[kk].d[4] = relu_pk_bf16(hi.x);
    bfrag[kk].d[5] = relu_pk_bf16(hi.y);
    bfrag[kk].d[6] = relu_pk_bf16(hi.z);
    bfrag[kk].d[7] = relu_pk_bf16(hi.w);
  }

#pragma unroll
  for (int MT = 0; MT < 8; ++MT) {
    const int row = MT * 16 + mrow;
    v8f c = {};
#pragma unroll
    for (int kk = 0; kk < 2; ++kk) {
      const uint4* ap = (const uint4*)(wlds + row * 64 + kk * 32 + half * 16);
      uint4 alo = ap[0], ahi = ap[1];
      Frag af;
      af.d[0] = alo.x; af.d[1] = alo.y; af.d[2] = alo.z; af.d[3] = alo.w;
      af.d[4] = ahi.x; af.d[5] = ahi.y; af.d[6] = ahi.z; af.d[7] = ahi.w;
      c = __builtin_amdgcn_wmma_f32_16x16x32_bf16(false, af.b, false, bfrag[kk].b,
                                                  (short)0, c, false, false);
    }
    const int fo = MT * 16 + half * 8;
    float s[8];
#pragma unroll
    for (int v = 0; v < 8; ++v) s[v] = c[v] + bias[g * 128 + fo + v];
    // reduce over the 8 columns (= 8 symmetries) sharing one batch index
#pragma unroll
    for (int m = 1; m < 8; m <<= 1)
#pragma unroll
      for (int v = 0; v < 8; ++v) s[v] += __shfl_xor(s[v], m, 32);
    if ((lane & 7) == 0) {
      long bidx = (colbase + wave * 16 + mrow) >> 3;
#pragma unroll
      for (int v = 0; v < 8; ++v) atomicAdd(b5 + bidx * 128 + fo + v, s[v]);
    }
  }
}

// ---------------------------------------------------------------------------
// Head MLP: b5 [B,128] -> BN/ReLU/Lin x2 -> BN/ReLU -> dot(fin) -> y [B]
// ---------------------------------------------------------------------------
__global__ __launch_bounds__(128) void head_kernel(
    const float* __restrict__ b5,
    const float* g0, const float* be0, const float* m0, const float* v0,
    const float* g1, const float* be1, const float* m1, const float* v1,
    const float* g2, const float* be2, const float* m2, const float* v2,
    const float* __restrict__ l0, const float* __restrict__ l1,
    const float* __restrict__ fw, const float* __restrict__ fb,
    float* __restrict__ out) {
  __shared__ float sh[128];
  const int b = blockIdx.x, t = threadIdx.x;
  float v = b5[(long)b * 128 + t];
  v = (v - m0[t]) * (g0[t] * rsqrtf(v0[t] + 1e-5f)) + be0[t];
  sh[t] = fmaxf(v, 0.f);
  __syncthreads();
  float acc = 0.f;
  for (int k = 0; k < 128; ++k) acc += sh[k] * l0[t * 128 + k];
  __syncthreads();
  v = (acc - m1[t]) * (g1[t] * rsqrtf(v1[t] + 1e-5f)) + be1[t];
  sh[t] = fmaxf(v, 0.f);
  __syncthreads();
  acc = 0.f;
  for (int k = 0; k < 128; ++k) acc += sh[k] * l1[t * 128 + k];
  __syncthreads();
  v = (acc - m2[t]) * (g2[t] * rsqrtf(v2[t] + 1e-5f)) + be2[t];
  sh[t] = fmaxf(v, 0.f) * fw[t];
  __syncthreads();
  for (int off = 64; off > 0; off >>= 1) {
    if (t < off) sh[t] += sh[t + off];
    __syncthreads();
  }
  if (t == 0) out[b] = sh[0] + fb[0];
}

// ---------------------------------------------------------------------------
extern "C" void kernel_launch(void* const* d_in, const int* in_sizes, int n_in,
                              void* d_out, int out_size, void* d_ws, size_t ws_size,
                              hipStream_t stream) {
  const float* x = (const float*)d_in[0];
  EmbedPtrs ep;
  for (int i = 0; i < 16; ++i) ep.w[i] = (const float*)d_in[1 + i];
  const float* mid_w[9]; for (int i = 0; i < 9; ++i) mid_w[i] = (const float*)d_in[17 + i];
  const float* mid_b[9]; for (int i = 0; i < 9; ++i) mid_b[i] = (const float*)d_in[26 + i];
  const float* last_w = (const float*)d_in[35];
  const float* last_b = (const float*)d_in[36];
  const float* bng[3] = {(const float*)d_in[37], (const float*)d_in[38], (const float*)d_in[39]};
  const float* bnb[3] = {(const float*)d_in[40], (const float*)d_in[41], (const float*)d_in[42]};
  const float* bnm[3] = {(const float*)d_in[43], (const float*)d_in[44], (const float*)d_in[45]};
  const float* bnv[3] = {(const float*)d_in[46], (const float*)d_in[47], (const float*)d_in[48]};
  const float* lin0 = (const float*)d_in[49];
  const float* lin1 = (const float*)d_in[50];
  const float* fin_w = (const float*)d_in[51];
  const float* fin_b = (const float*)d_in[52];

  char* ws = (char*)d_ws;
  const size_t ACT = (size_t)NGRP * NCOLS * 64 * 2;   // 134,217,728 B
  __bf16* A0 = (__bf16*)(ws);                          // m / residual base
  __bf16* A1 = (__bf16*)(ws + ACT);
  __bf16* A2 = (__bf16*)(ws + 2 * ACT);
  unsigned short* WB = (unsigned short*)(ws + 3 * ACT); // bf16 weights
  const size_t WELEMS = 9 * 65536 + 131072;             // mid + last
  float* B5 = (float*)(ws + 3 * ACT + WELEMS * 2);      // [8192][128]

  // 1) convert conv weights to bf16
  for (int i = 0; i < 9; ++i)
    cvt_bf16_kernel<<<256, 256, 0, stream>>>(mid_w[i], WB + (size_t)i * 65536, 65536);
  cvt_bf16_kernel<<<256, 256, 0, stream>>>(last_w, WB + (size_t)9 * 65536, 131072);

  // 2) zero b5 accumulator
  hipMemsetAsync(B5, 0, (size_t)8192 * 128 * sizeof(float), stream);

  // 3) symmetry expansion + pattern embedding -> A0 (= m)
  embed_kernel<<<8192, 256, 0, stream>>>(x, ep, A0);

  // 4) three residual blocks of 3 grouped convs each
  dim3 grid(NCOLS / 128, NGRP);
  for (int blk = 0; blk < 3; ++blk) {
    const __bf16* w0 = (const __bf16*)(WB + (size_t)(3 * blk + 0) * 65536);
    const __bf16* w1 = (const __bf16*)(WB + (size_t)(3 * blk + 1) * 65536);
    const __bf16* w2 = (const __bf16*)(WB + (size_t)(3 * blk + 2) * 65536);
    conv64_wmma_kernel<<<grid, 256, 0, stream>>>(A0, A1, w0, mid_b[3 * blk + 0], nullptr);
    conv64_wmma_kernel<<<grid, 256, 0, stream>>>(A1, A2, w1, mid_b[3 * blk + 1], nullptr);
    conv64_wmma_kernel<<<grid, 256, 0, stream>>>(A2, A0, w2, mid_b[3 * blk + 2], A0);
  }

  // 5) last conv (128 out) fused with sum over groups & symmetries
  lastconv_wmma_kernel<<<grid, 256, 0, stream>>>(
      A0, (const __bf16*)(WB + (size_t)9 * 65536), last_b, B5);

  // 6) head MLP
  head_kernel<<<8192, 128, 0, stream>>>(
      B5, bng[0], bnb[0], bnm[0], bnv[0], bng[1], bnb[1], bnm[1], bnv[1],
      bng[2], bnb[2], bnm[2], bnv[2], lin0, lin1, fin_w, fin_b, (float*)d_out);
}
